// RetNetBlock_87076166960155
// MI455X (gfx1250) — compile-verified, matrix operand order checked by reading, not run on hardware
//
#include <hip/hip_runtime.h>

// ---------------------------------------------------------------------------
// RetNet block for MI455X (gfx1250): bf16 WMMA GEMMs with async-LDS double
// buffering + register-resident retention scan.
// B=2, T=2048, D=1024, H=16, d=64, I=2816
// ---------------------------------------------------------------------------

typedef __attribute__((ext_vector_type(16))) __bf16 v16bf;
typedef __attribute__((ext_vector_type(8)))  __bf16 v8bf;
typedef __attribute__((ext_vector_type(8)))  float  v8f;
typedef int v4i __attribute__((vector_size(16)));   // b128 payload type

#define B_ 2
#define T_ 2048
#define D_ 1024
#define H_ 16
#define dh_ 64
#define I_ 2816
#define M_ (B_ * T_)   // 4096 token rows

// --------------------- CDNA5 async global->LDS copy -----------------------
#if defined(__gfx1250__) && __has_builtin(__builtin_amdgcn_global_load_async_to_lds_b128) && \
    __has_builtin(__builtin_amdgcn_s_wait_asynccnt)
#define ASYNC_LDS 1
#endif

#define AS1 __attribute__((address_space(1)))
#define AS3 __attribute__((address_space(3)))

__device__ __forceinline__ void ldg_async_b128(void* lds, const void* gptr) {
#ifdef ASYNC_LDS
  // generic->AS roundtrip through integers: LDS offset = low 32 bits of the
  // generic address (gfx1250 aperture rule); global generic addr is identity.
  // builtin prototype: void(v4i32 AS1*, v4i32 AS3*, imm offset, imm cpol)
  __builtin_amdgcn_global_load_async_to_lds_b128(
      (AS1 v4i*)(unsigned long long)(uintptr_t)gptr,
      (AS3 v4i*)(unsigned)(uintptr_t)lds, 0, 0);
#else
  *(v8bf*)lds = *(const v8bf*)gptr;
#endif
}

__device__ __forceinline__ void async_wait_all() {
#ifdef ASYNC_LDS
  __builtin_amdgcn_s_wait_asynccnt(0);
#endif
}

// ------------------------------- converts ---------------------------------
__global__ __launch_bounds__(256) void f32_to_bf16_kernel(const float* __restrict__ in,
                                                          __bf16* __restrict__ out, int n) {
  for (int i = blockIdx.x * blockDim.x + threadIdx.x; i < n; i += gridDim.x * blockDim.x)
    out[i] = (__bf16)in[i];
}

// ------------------------------- RMSNorm ----------------------------------
__global__ __launch_bounds__(256) void rmsnorm_kernel(const float* __restrict__ x,
                                                      const float* __restrict__ w,
                                                      __bf16* __restrict__ out) {
  __shared__ float red[256];
  const int row = blockIdx.x;
  const int tid = threadIdx.x;
  const float* xr = x + (size_t)row * D_;
  float4 v = *(const float4*)(xr + tid * 4);
  float ss = v.x * v.x + v.y * v.y + v.z * v.z + v.w * v.w;
  red[tid] = ss;
  __syncthreads();
  for (int s = 128; s > 0; s >>= 1) {
    if (tid < s) red[tid] += red[tid + s];
    __syncthreads();
  }
  const float scale = rsqrtf(red[0] * (1.0f / D_) + 1e-6f);
  __bf16* orow = out + (size_t)row * D_;
  const float4 wv = *(const float4*)(w + tid * 4);
  orow[tid * 4 + 0] = (__bf16)(v.x * scale * wv.x);
  orow[tid * 4 + 1] = (__bf16)(v.y * scale * wv.y);
  orow[tid * 4 + 2] = (__bf16)(v.z * scale * wv.z);
  orow[tid * 4 + 3] = (__bf16)(v.w * scale * wv.w);
}

// ------------------------------ WMMA GEMM ---------------------------------
// out[M,N] = A[M,K] (bf16, row-major) x W[N,K]^T (bf16, row-major) + epilogue
// 256 threads = 8 waves; workgroup tile 128x128; wave tile 64x32 (4x2 frags)
// Async-LDS double-buffered pipeline: stage tile k+1 while computing tile k.
// EPI: 0 = store bf16; 1 = sigmoid->bf16; 2 = silu->bf16;
//      3 = (acc * mulb)->bf16; 4 = (acc + resid)->f32
template <int EPI>
__global__ __launch_bounds__(256) void wmma_gemm_kernel(
    const __bf16* __restrict__ A, const __bf16* __restrict__ W,
    __bf16* __restrict__ outb, float* __restrict__ outf,
    const float* __restrict__ resid, const __bf16* __restrict__ mulb,
    int M, int N, int K) {
  constexpr int BM = 128, BN = 128, BK = 64, LDT = 72;  // +8 bf16 pad
  __shared__ __align__(16) __bf16 sA[2][BM * LDT];
  __shared__ __align__(16) __bf16 sB[2][BN * LDT];

  const int tid  = threadIdx.x;
  const int lane = tid & 31;
  const int wid  = tid >> 5;
  const int wm   = wid >> 2;      // 0..1  (M dir)
  const int wn   = wid & 3;       // 0..3  (N dir)
  const int lm   = lane & 15;
  const int half = lane >> 4;
  const int m0 = blockIdx.y * BM;
  const int n0 = blockIdx.x * BN;

  v8f acc[4][2] = {};

  // stage one 128x64 bf16 tile pair into LDS buffer `buf`
  auto stage = [&](int buf, int k0) {
#pragma unroll
    for (int it = 0; it < 4; ++it) {
      int lin = (tid + it * 256) * 8;  // 0..8191
      int r = lin >> 6, c = lin & 63;
      ldg_async_b128(&sA[buf][r * LDT + c], &A[(size_t)(m0 + r) * K + k0 + c]);
      ldg_async_b128(&sB[buf][r * LDT + c], &W[(size_t)(n0 + r) * K + k0 + c]);
    }
  };

  stage(0, 0);
  async_wait_all();
  __syncthreads();

  int buf = 0;
  for (int k0 = 0; k0 < K; k0 += BK) {
    if (k0 + BK < K) stage(buf ^ 1, k0 + BK);  // overlap copy with compute

    const __bf16* pA = &sA[buf][0];
    const __bf16* pB = &sB[buf][0];
#pragma unroll
    for (int kk = 0; kk < BK; kk += 32) {
      union Frag { v16bf v; v8bf h[2]; };
      v16bf a[4], b[2];
#pragma unroll
      for (int i = 0; i < 4; ++i) {
        const __bf16* p = &pA[(wm * 64 + i * 16 + lm) * LDT + kk + half * 8];
        Frag f; f.h[0] = *(const v8bf*)p; f.h[1] = *(const v8bf*)(p + 16);
        a[i] = f.v;
      }
#pragma unroll
      for (int j = 0; j < 2; ++j) {
        const __bf16* p = &pB[(wn * 32 + j * 16 + lm) * LDT + kk + half * 8];
        Frag f; f.h[0] = *(const v8bf*)p; f.h[1] = *(const v8bf*)(p + 16);
        b[j] = f.v;
      }
#pragma unroll
      for (int i = 0; i < 4; ++i)
#pragma unroll
        for (int j = 0; j < 2; ++j)
          acc[i][j] = __builtin_amdgcn_wmma_f32_16x16x32_bf16(
              false, a[i], false, b[j], (short)0, acc[i][j], false, false);
    }
    async_wait_all();   // our async copies for the next tile are done
    __syncthreads();    // everyone finished reading `buf` + staging `buf^1`
    buf ^= 1;
  }

  // epilogue: C/D layout -> lane<16: (M=r, N=lane); lane>=16: (M=8+r, N=lane-16)
#pragma unroll
  for (int i = 0; i < 4; ++i)
#pragma unroll
    for (int j = 0; j < 2; ++j)
#pragma unroll
      for (int r = 0; r < 8; ++r) {
        const int mrow = m0 + wm * 64 + i * 16 + r + half * 8;
        const int ncol = n0 + wn * 32 + j * 16 + lm;
        const size_t o = (size_t)mrow * N + ncol;
        float v = acc[i][j][r];
        if constexpr (EPI == 0) {
          outb[o] = (__bf16)v;
        } else if constexpr (EPI == 1) {
          outb[o] = (__bf16)(1.0f / (1.0f + __expf(-v)));
        } else if constexpr (EPI == 2) {
          outb[o] = (__bf16)(v / (1.0f + __expf(-v)));
        } else if constexpr (EPI == 3) {
          outb[o] = (__bf16)(v * (float)mulb[o]);
        } else {
          outf[o] = v + resid[o];
        }
      }
}

// ---------------------------- retention scan ------------------------------
// one block per (b,h). 64x64 state in registers: 256 threads x 16 f32.
// thread (rg = tid/64, c = tid%64) owns S[rg*16 .. rg*16+15][c].
// fuses GroupNorm sum/sumsq over the (T,d) extent.
__global__ __launch_bounds__(256) void retnet_scan_kernel(
    const __bf16* __restrict__ qb, const __bf16* __restrict__ kb,
    const __bf16* __restrict__ vb, const float* __restrict__ gamma,
    float* __restrict__ y, float* __restrict__ stats) {
  const int bh = blockIdx.x;
  const int b = bh >> 4, h = bh & 15;
  const float gam = 1.0f / (1.0f + __expf(-gamma[h]));
  const int tid = threadIdx.x;
  const int c = tid & 63, rg = tid >> 6, rbase = rg * 16;

  __shared__ float sq[64], sk[64], sv[64];
  __shared__ float part[4][64];
  __shared__ float redA[64], redB[64];

  float S[16];
#pragma unroll
  for (int i = 0; i < 16; ++i) S[i] = 0.0f;
  float sum = 0.0f, sumsq = 0.0f;

  const size_t baseRow = (size_t)b * T_ * D_ + h * dh_;
  for (int t = 0; t < T_; ++t) {
    const size_t a = baseRow + (size_t)t * D_;
    if (tid < 64)       sq[tid]       = (float)qb[a + tid];
    else if (tid < 128) sk[tid - 64]  = (float)kb[a + tid - 64];
    else if (tid < 192) sv[tid - 128] = (float)vb[a + tid - 128];
    __syncthreads();
    const float vv = sv[c];
    float accp = 0.0f;
#pragma unroll
    for (int i = 0; i < 16; ++i) {
      S[i] = gam * S[i] + sk[rbase + i] * vv;   // S = gamma*S + k v^T
      accp += sq[rbase + i] * S[i];             // o = q . S
    }
    part[rg][c] = accp;
    __syncthreads();
    if (rg == 0) {
      const float o = part[0][c] + part[1][c] + part[2][c] + part[3][c];
      y[a + c] = o;
      sum += o;
      sumsq += o * o;
    }
    __syncthreads();
  }
  if (rg == 0) { redA[c] = sum; redB[c] = sumsq; }
  __syncthreads();
  if (tid == 0) {
    float s = 0.0f, s2 = 0.0f;
    for (int i = 0; i < 64; ++i) { s += redA[i]; s2 += redB[i]; }
    const float n = (float)T_ * 64.0f;
    const float mean = s / n;
    const float var = s2 / n - mean * mean;
    stats[bh * 2 + 0] = mean;
    stats[bh * 2 + 1] = rsqrtf(var + 1e-5f);
  }
}

// --------------------- gated GroupNorm (elementwise) ----------------------
__global__ __launch_bounds__(256) void gn_gate_kernel(
    const float* __restrict__ y, const __bf16* __restrict__ gb,
    const float* __restrict__ stats, const float* __restrict__ gn_w,
    const float* __restrict__ gn_b, __bf16* __restrict__ out) {
  const size_t i = (size_t)blockIdx.x * blockDim.x + threadIdx.x;
  const int col = (int)(i & (D_ - 1));
  const int h = col >> 6;
  const int row = (int)(i >> 10);
  const int b = row >> 11;  // row / T_
  const int sidx = (b * H_ + h) * 2;
  const float mean = stats[sidx], rstd = stats[sidx + 1];
  const float val = (y[i] - mean) * rstd * gn_w[col] + gn_b[col];
  out[i] = (__bf16)((float)gb[i] * val);
}

// ------------------------------- launcher ---------------------------------
static inline size_t align256(size_t x) { return (x + 255) & ~(size_t)255; }

extern "C" void kernel_launch(void* const* d_in, const int* in_sizes, int n_in,
                              void* d_out, int out_size, void* d_ws, size_t ws_size,
                              hipStream_t stream) {
  const float* x      = (const float*)d_in[0];
  const float* ln1_w  = (const float*)d_in[1];
  const float* ln2_w  = (const float*)d_in[2];
  const float* Wq     = (const float*)d_in[3];
  const float* Wk     = (const float*)d_in[4];
  const float* Wv     = (const float*)d_in[5];
  const float* Wg     = (const float*)d_in[6];
  const float* Wo     = (const float*)d_in[7];
  const float* gamma  = (const float*)d_in[8];
  const float* gn_w   = (const float*)d_in[9];
  const float* gn_b   = (const float*)d_in[10];
  const float* gate_w = (const float*)d_in[11];
  const float* up_w   = (const float*)d_in[12];
  const float* down_w = (const float*)d_in[13];
  float* out = (float*)d_out;

  char* ws = (char*)d_ws;
  size_t cur = 0;
  auto alloc = [&](size_t bytes) { void* p = ws + cur; cur = align256(cur + bytes); return p; };

  __bf16* Wqb  = (__bf16*)alloc((size_t)D_ * D_ * 2);
  __bf16* Wkb  = (__bf16*)alloc((size_t)D_ * D_ * 2);
  __bf16* Wvb  = (__bf16*)alloc((size_t)D_ * D_ * 2);
  __bf16* Wgb  = (__bf16*)alloc((size_t)D_ * D_ * 2);
  __bf16* Wob  = (__bf16*)alloc((size_t)D_ * D_ * 2);
  __bf16* gwb  = (__bf16*)alloc((size_t)I_ * D_ * 2);
  __bf16* uwb  = (__bf16*)alloc((size_t)I_ * D_ * 2);
  __bf16* dwb  = (__bf16*)alloc((size_t)D_ * I_ * 2);
  __bf16* hb   = (__bf16*)alloc((size_t)M_ * D_ * 2);
  __bf16* qb   = (__bf16*)alloc((size_t)M_ * D_ * 2);
  __bf16* kb   = (__bf16*)alloc((size_t)M_ * D_ * 2);
  __bf16* vb   = (__bf16*)alloc((size_t)M_ * D_ * 2);
  __bf16* gb   = (__bf16*)alloc((size_t)M_ * D_ * 2);
  float*  yf   = (float*) alloc((size_t)M_ * D_ * 4);
  float*  stats= (float*) alloc((size_t)B_ * H_ * 2 * 4);
  __bf16* y2b  = (__bf16*)alloc((size_t)M_ * D_ * 2);
  float*  x1   = (float*) alloc((size_t)M_ * D_ * 4);
  __bf16* h2b  = (__bf16*)alloc((size_t)M_ * D_ * 2);
  __bf16* gact = (__bf16*)alloc((size_t)M_ * I_ * 2);
  __bf16* midb = (__bf16*)alloc((size_t)M_ * I_ * 2);
  (void)ws_size; (void)n_in; (void)in_sizes; (void)out_size;

  const dim3 blk(256);
  // 1. weights -> bf16
  f32_to_bf16_kernel<<<2048, blk, 0, stream>>>(Wq, Wqb, D_ * D_);
  f32_to_bf16_kernel<<<2048, blk, 0, stream>>>(Wk, Wkb, D_ * D_);
  f32_to_bf16_kernel<<<2048, blk, 0, stream>>>(Wv, Wvb, D_ * D_);
  f32_to_bf16_kernel<<<2048, blk, 0, stream>>>(Wg, Wgb, D_ * D_);
  f32_to_bf16_kernel<<<2048, blk, 0, stream>>>(Wo, Wob, D_ * D_);
  f32_to_bf16_kernel<<<2048, blk, 0, stream>>>(gate_w, gwb, I_ * D_);
  f32_to_bf16_kernel<<<2048, blk, 0, stream>>>(up_w, uwb, I_ * D_);
  f32_to_bf16_kernel<<<2048, blk, 0, stream>>>(down_w, dwb, D_ * I_);

  // 2. h = rmsnorm(x) -> bf16
  rmsnorm_kernel<<<M_, blk, 0, stream>>>(x, ln1_w, hb);

  // 3. q,k,v,g projections (WMMA bf16)
  dim3 gDD(D_ / 128, M_ / 128);
  wmma_gemm_kernel<0><<<gDD, blk, 0, stream>>>(hb, Wqb, qb, nullptr, nullptr, nullptr, M_, D_, D_);
  wmma_gemm_kernel<0><<<gDD, blk, 0, stream>>>(hb, Wkb, kb, nullptr, nullptr, nullptr, M_, D_, D_);
  wmma_gemm_kernel<0><<<gDD, blk, 0, stream>>>(hb, Wvb, vb, nullptr, nullptr, nullptr, M_, D_, D_);
  wmma_gemm_kernel<1><<<gDD, blk, 0, stream>>>(hb, Wgb, gb, nullptr, nullptr, nullptr, M_, D_, D_);

  // 4. retention scan (+ fused GroupNorm statistics)
  retnet_scan_kernel<<<B_ * H_, blk, 0, stream>>>(qb, kb, vb, gamma, yf, stats);

  // 5. y2 = g * GroupNorm(y) -> bf16
  gn_gate_kernel<<<(M_ * D_) / 256, blk, 0, stream>>>(yf, gb, stats, gn_w, gn_b, y2b);

  // 6. x1 = x + y2 @ Wo^T  (f32 residual epilogue)
  wmma_gemm_kernel<4><<<gDD, blk, 0, stream>>>(y2b, Wob, nullptr, x1, x, nullptr, M_, D_, D_);

  // 7. h2 = rmsnorm(x1) -> bf16
  rmsnorm_kernel<<<M_, blk, 0, stream>>>(x1, ln2_w, h2b);

  // 8. MLP: silu(h2@gate^T), (h2@up^T)*gate_act, then down + residual
  dim3 gID(I_ / 128, M_ / 128);
  wmma_gemm_kernel<2><<<gID, blk, 0, stream>>>(h2b, gwb, gact, nullptr, nullptr, nullptr, M_, I_, D_);
  wmma_gemm_kernel<3><<<gID, blk, 0, stream>>>(h2b, uwb, midb, nullptr, nullptr, gact, M_, I_, D_);
  dim3 gDI(D_ / 128, M_ / 128);
  wmma_gemm_kernel<4><<<gDI, blk, 0, stream>>>(midb, dwb, nullptr, out, x1, nullptr, M_, D_, I_);
}
